// S4DKernel_49563922596386
// MI455X (gfx1250) — matrix operand order repaired and checked by reading, not compiled, not run on hardware
//
#include <hip/hip_runtime.h>
#include <hip/hip_bf16.h>
#include <cmath>

typedef __attribute__((ext_vector_type(16))) _Float16 v16h;
typedef __attribute__((ext_vector_type(8)))  float    v8f;

#define D_MODEL 1024
#define D_STATE 64
#define KDIM    128                    // [CB_re | -CB_im] stacked
#define NT      16                     // l-tile width (WMMA N)
#define WAVES_PER_BLOCK 8
#define THREADS (WAVES_PER_BLOCK * 32)
#define MTILES  (D_MODEL / 16)         // 64
#define MT_PER_WAVE (MTILES / WAVES_PER_BLOCK)  // 8

// ---------------------------------------------------------------------------
// Prep 1: log_w[n] = log(exp(A)) with imag wrapped to (-pi,pi] (atan2), per ref.
// logw[0..63] = Re, logw[64..127] = Im.
// ---------------------------------------------------------------------------
__global__ void s4d_logw_kernel(const float* __restrict__ A_log,
                                const float* __restrict__ A_imag,
                                float* __restrict__ logw) {
    int n = threadIdx.x;
    if (n < D_STATE) {
        float are = -expf(A_log[n]);
        float ai  = A_imag[n];
        float cr  = cosf(ai);
        float si  = sinf(ai);
        float mag = expf(are) * sqrtf(cr * cr + si * si);
        logw[n]            = logf(mag);
        logw[D_STATE + n]  = atan2f(si, cr);
    }
}

// ---------------------------------------------------------------------------
// Prep 2: pack A = [2*CB_re | -2*CB_im] (1024 x 128 f16) directly into the
// CDNA5 16-bit A-matrix fragment layout (ISA 7.12.2):
//   lane 0-15:  M = lane,   halves 0-7 -> K 0-7,  halves 8-15 -> K 16-23
//   lane 16-31: M = lane-16,halves 0-7 -> K 8-15, halves 8-15 -> K 24-31
// Flat layout: Apack[((t*4 + kb)*32 + lane)*16 + h], t = M-tile, kb = K-block.
// ---------------------------------------------------------------------------
__global__ void s4d_packA_kernel(const float* __restrict__ B_re,
                                 const float* __restrict__ B_im,
                                 const float* __restrict__ C_re,
                                 const float* __restrict__ C_im,
                                 _Float16* __restrict__ Apack) {
    int f = blockIdx.x * blockDim.x + threadIdx.x;   // 0 .. 1024*128-1
    if (f >= D_MODEL * KDIM) return;
    int t    = f >> 11;          // / 2048
    int rem  = f & 2047;
    int kb   = rem >> 9;         // / 512
    int lane = (rem >> 4) & 31;
    int h    = rem & 15;

    int row = lane & 15;
    int M   = t * 16 + row;
    int k_local = (h < 8 ? h : h + 8) + ((lane >= 16) ? 8 : 0);
    int K   = kb * 32 + k_local;

    float v;
    if (K < D_STATE) {
        int n = K;
        v =  2.0f * (C_re[M * D_STATE + n] * B_re[M * D_STATE + n] -
                     C_im[M * D_STATE + n] * B_im[M * D_STATE + n]);
    } else {
        int n = K - D_STATE;
        v = -2.0f * (C_re[M * D_STATE + n] * B_im[M * D_STATE + n] +
                     C_im[M * D_STATE + n] * B_re[M * D_STATE + n]);
    }
    Apack[f] = (_Float16)v;
}

// ---------------------------------------------------------------------------
// Main: one block per 16-wide l-tile. Build the 128x16 B tile (V_re over
// V_im) analytically, in fragment layout, shared in LDS. Each wave then runs
// the M loop: 4 chained v_wmma_f32_16x16x32_f16 per 16x16 output tile.
// Output stores are NON-TEMPORAL: 256 MB write-once stream vs 192 MB L2 —
// keep it out of L2 so the shared Apack/logw stay resident.
// ---------------------------------------------------------------------------
__global__ __launch_bounds__(THREADS) void s4d_wmma_kernel(
    const _Float16* __restrict__ Apack,
    const float*    __restrict__ logw,
    float*          __restrict__ out,
    int L) {
    __shared__ __align__(32) _Float16 Bsh[KDIM * NT];   // 2048 halves, frag layout

    const int tid = threadIdx.x;
    const int l0  = blockIdx.x * NT;

    // Cooperative B-tile build: each thread fills 8 contiguous halves.
    #pragma unroll
    for (int i = 0; i < 8; ++i) {
        int f    = tid * 8 + i;
        int kb   = f >> 9;            // / 512
        int lane = (f >> 4) & 31;
        int h    = f & 15;
        int k_local = (h < 8 ? h : h + 8) + ((lane >= 16) ? 8 : 0);
        int K    = kb * 32 + k_local;
        int col  = lane & 15;
        float l  = (float)(l0 + col);

        int   n  = (K < D_STATE) ? K : (K - D_STATE);
        float lr = logw[n];
        float li = logw[D_STATE + n];
        float decay = expf(lr * l);               // underflows to 0 naturally
        float ang   = li * l;
        float v = decay * ((K < D_STATE) ? cosf(ang) : sinf(ang));
        Bsh[f] = (_Float16)v;
    }
    __syncthreads();

    const int wave = tid >> 5;
    const int lane = tid & 31;

    v16h b[4];
    #pragma unroll
    for (int kb = 0; kb < 4; ++kb)
        b[kb] = *(const v16h*)(&Bsh[(kb * 32 + lane) * 16]);

    #pragma unroll
    for (int mt = 0; mt < MT_PER_WAVE; ++mt) {
        int t = wave + mt * WAVES_PER_BLOCK;
        const _Float16* ap = Apack + ((size_t)t * 4 * 32) * 16;

        if (mt + 1 < MT_PER_WAVE) {
            const _Float16* nxt =
                Apack + ((size_t)(t + WAVES_PER_BLOCK) * 4 * 32) * 16 + lane * 16;
            __builtin_prefetch(nxt, 0, 1);        // global_prefetch_b8
        }

        v8f c = {};
        #pragma unroll
        for (int kb = 0; kb < 4; ++kb) {
            v16h a = *(const v16h*)(ap + ((size_t)(kb * 32 + lane)) * 16);
            c = __builtin_amdgcn_wmma_f32_16x16x32_f16(
                    /*neg_a=*/false, a, /*neg_b=*/false, b[kb],
                    /*c_mod=*/(short)0, c, /*reuse_a=*/false, /*reuse_b=*/false);
        }

        // C/D layout: VGPR r -> M = r (lanes 0-15) / r+8 (lanes 16-31), N = lane%16
        int rowBase = t * 16 + ((lane >= 16) ? 8 : 0);
        int col     = l0 + (lane & 15);
        if (col < L) {
            float* o = out + (size_t)rowBase * L + col;
            #pragma unroll
            for (int r = 0; r < 8; ++r)
                __builtin_nontemporal_store(c[r], o + (size_t)r * L);
        }
    }
}

// ---------------------------------------------------------------------------
extern "C" void kernel_launch(void* const* d_in, const int* in_sizes, int n_in,
                              void* d_out, int out_size, void* d_ws, size_t ws_size,
                              hipStream_t stream) {
    const float* A_log  = (const float*)d_in[0];
    const float* A_imag = (const float*)d_in[1];
    const float* B_re   = (const float*)d_in[2];
    const float* B_im   = (const float*)d_in[3];
    const float* C_re   = (const float*)d_in[4];
    const float* C_im   = (const float*)d_in[5];
    float* out = (float*)d_out;

    // L is a device-resident scalar, but out_size = D_MODEL * L.
    const int L = out_size / D_MODEL;

    // Workspace layout: [Apack: 1024*128 f16 = 256KB][logw: 128 f32]
    _Float16* Apack = (_Float16*)d_ws;
    float*    logw  = (float*)((char*)d_ws + (size_t)D_MODEL * KDIM * sizeof(_Float16));

    s4d_logw_kernel<<<1, 64, 0, stream>>>(A_log, A_imag, logw);
    s4d_packA_kernel<<<(D_MODEL * KDIM + 255) / 256, 256, 0, stream>>>(
        B_re, B_im, C_re, C_im, Apack);

    const int nblocks = (L + NT - 1) / NT;   // 4096 for L = 65536
    s4d_wmma_kernel<<<nblocks, THREADS, 0, stream>>>(Apack, logw, out, L);
}
